// ALSTM_MUL_6622839570830
// MI455X (gfx1250) — matrix operand order repaired and checked by reading, not compiled, not run on hardware
//
#include <hip/hip_runtime.h>
#include <stdint.h>

#define T_DIM 128
#define B_DIM 128
#define I_DIM 128
#define H_DIM 256
#define FC_DIM 32

typedef __bf16 v16bf __attribute__((ext_vector_type(16)));
typedef float  v8f   __attribute__((ext_vector_type(8)));
typedef unsigned int u32x4 __attribute__((ext_vector_type(4)));

__device__ __forceinline__ unsigned short f2bf(float f) {
  unsigned int u = __float_as_uint(f);
  u += 0x7fffu + ((u >> 16) & 1u);          // round-to-nearest-even
  return (unsigned short)(u >> 16);
}
__device__ __forceinline__ float bflo(unsigned int u) { return __uint_as_float(u << 16); }
__device__ __forceinline__ float bfhi(unsigned int u) { return __uint_as_float(u & 0xffff0000u); }
__device__ __forceinline__ float fsig(float x)  { return 1.f / (1.f + __expf(-x)); }
__device__ __forceinline__ float ftanh(float x) { return 1.f - 2.f / (__expf(2.f * x) + 1.f); }

// ---------------- f32 -> bf16 convert ----------------
__global__ void cvt_bf16_kernel(const float* __restrict__ in,
                                unsigned short* __restrict__ out, int n) {
  int i = blockIdx.x * blockDim.x + threadIdx.x;
  int stride = gridDim.x * blockDim.x;
  for (; i < n; i += stride) out[i] = f2bf(in[i]);
}

// ---------------- batched WMMA projection GEMM ----------------
// P[gate][b][t][h] = sum_k A(b,t,k) * W[gate][b][h][k]
// A row for (b,t): A + b*a_boff + t*a_tstride (k-contiguous, bf16)
// W: [4][B][H][K] bf16 row-major (k-contiguous) -> used directly as B-matrix (W^T)
__global__ __launch_bounds__(256)
void proj_wmma_kernel(const unsigned short* __restrict__ A, long long a_boff, long long a_tstride,
                      const unsigned short* __restrict__ W, int K,
                      float* __restrict__ P) {
  const int lane = threadIdx.x & 31;
  const int wave = (blockIdx.x << 3) + (threadIdx.x >> 5);   // 16384 waves total
  const int ngroup = wave & 3;              // 4 groups of 4 n-tiles (H=256 -> 16 n-tiles)
  const int ttile  = (wave >> 2) & 7;       // T=128 -> 8 m-tiles
  const int b      = (wave >> 5) & (B_DIM - 1);
  const int gate   = wave >> 12;            // 0..3

  const int arow  = ttile * 16 + (lane & 15);
  const int kbase = (lane >> 4) << 3;       // ISA 16-bit A layout: lanes16-31 start at K=8
  const unsigned short* Arow = A + (long long)b * a_boff + (long long)arow * a_tstride;

  const unsigned short* Wg = W + ((long long)(gate * B_DIM + b)) * H_DIM * (long long)K;
  const int kb_lane = (lane >> 4) << 3;     // ISA 16-bit B layout: lanes16-31 start at K=8 (same as A)

  v8f acc[4];
#pragma unroll
  for (int nt = 0; nt < 4; ++nt)
#pragma unroll
    for (int r = 0; r < 8; ++r) acc[nt][r] = 0.f;

  const int KK = K >> 5;
  for (int kk = 0; kk < KK; ++kk) {
    union { v16bf v; u32x4 q[2]; } ua;
    const int k0 = (kk << 5) + kbase;
    ua.q[0] = *(const u32x4*)(Arow + k0);        // local K {0..7}/{8..15}
    ua.q[1] = *(const u32x4*)(Arow + k0 + 16);   // local K {16..23}/{24..31}
    const int kb = (kk << 5) + kb_lane;
#pragma unroll
    for (int nt = 0; nt < 4; ++nt) {
      const int n = (ngroup << 6) + (nt << 4) + (lane & 15);
      const unsigned short* wp = Wg + (long long)n * K + kb;
      union { v16bf v; u32x4 q[2]; } ub;
      ub.q[0] = *(const u32x4*)(wp);
      ub.q[1] = *(const u32x4*)(wp + 16);
      acc[nt] = __builtin_amdgcn_wmma_f32_16x16x32_bf16(
          false, ua.v, false, ub.v, (short)0, acc[nt], false, false);
    }
  }

  float* Pg = P + ((long long)(gate * B_DIM + b)) * T_DIM * H_DIM;
  const int mrow0 = ttile * 16 + ((lane >> 4) << 3);   // lanes16-31 hold M=8..15
#pragma unroll
  for (int nt = 0; nt < 4; ++nt) {
    const int n = (ngroup << 6) + (nt << 4) + (lane & 15);
#pragma unroll
    for (int r = 0; r < 8; ++r)
      Pg[(long long)(mrow0 + r) * H_DIM + n] = acc[nt][r];
  }
}

// ---------------- recurrent scan (one workgroup per sample) ----------------
// P: [4][B][T][H] f32 preactivations; Wh: [4][B][H][H] bf16 (L2-resident)
__global__ __launch_bounds__(256)
void lstm_scan_kernel(const float* __restrict__ P,
                      const unsigned short* __restrict__ Wh,
                      const float* __restrict__ bgp, const float* __restrict__ bip,
                      const float* __restrict__ bfp, const float* __restrict__ bop,
                      unsigned short* __restrict__ hs_out,   // [B][T][H] bf16
                      float* __restrict__ hlast) {           // [B][H] f32
  __shared__ float h_lds[H_DIM];
  const int b = blockIdx.x;
  const int tid = threadIdx.x;
  const long long GSTRIDE = (long long)B_DIM * T_DIM * H_DIM;

  const float biasg = bgp[b * H_DIM + tid];
  const float biasi = bip[b * H_DIM + tid];
  const float biasf = bfp[b * H_DIM + tid];
  const float biaso = bop[b * H_DIM + tid];

  const unsigned short* w0 = Wh + ((long long)(0 * B_DIM + b) * H_DIM + tid) * H_DIM;
  const unsigned short* w1 = Wh + ((long long)(1 * B_DIM + b) * H_DIM + tid) * H_DIM;
  const unsigned short* w2 = Wh + ((long long)(2 * B_DIM + b) * H_DIM + tid) * H_DIM;
  const unsigned short* w3 = Wh + ((long long)(3 * B_DIM + b) * H_DIM + tid) * H_DIM;

  const float* Pb = P + (long long)b * T_DIM * H_DIM + tid;

  float c = 0.f;
  h_lds[tid] = 0.f;

  for (int t = 0; t < T_DIM; ++t) {
    __syncthreads();                          // h_lds stable for reads
    const long long po = (long long)t * H_DIM;
    float a0 = Pb[po + 0 * GSTRIDE] + biasg;
    float a1 = Pb[po + 1 * GSTRIDE] + biasi;
    float a2 = Pb[po + 2 * GSTRIDE] + biasf;
    float a3 = Pb[po + 3 * GSTRIDE] + biaso;

#pragma unroll 4
    for (int k = 0; k < H_DIM; k += 8) {
      u32x4 q0 = *(const u32x4*)(w0 + k);
      u32x4 q1 = *(const u32x4*)(w1 + k);
      u32x4 q2 = *(const u32x4*)(w2 + k);
      u32x4 q3 = *(const u32x4*)(w3 + k);
#pragma unroll
      for (int j = 0; j < 4; ++j) {
        const float he = h_lds[k + 2 * j];
        const float ho = h_lds[k + 2 * j + 1];
        a0 = fmaf(bflo(q0[j]), he, a0); a0 = fmaf(bfhi(q0[j]), ho, a0);
        a1 = fmaf(bflo(q1[j]), he, a1); a1 = fmaf(bfhi(q1[j]), ho, a1);
        a2 = fmaf(bflo(q2[j]), he, a2); a2 = fmaf(bfhi(q2[j]), ho, a2);
        a3 = fmaf(bflo(q3[j]), he, a3); a3 = fmaf(bfhi(q3[j]), ho, a3);
      }
    }

    const float gg = ftanh(a0);
    const float ii = fsig(a1);
    const float ff = fsig(a2);
    const float oo = fsig(a3);
    c = ff * c + ii * gg;
    const float hn = oo * ftanh(c);

    __syncthreads();                          // all reads done before update
    h_lds[tid] = hn;
    hs_out[((long long)b * T_DIM + t) * H_DIM + tid] = f2bf(hn);
    if (t == T_DIM - 1) hlast[b * H_DIM + tid] = hn;
  }
}

// ---------------- classifier head ----------------
__global__ void head_kernel(const float* __restrict__ hlast,
                            const float* __restrict__ fc1w, const float* __restrict__ fc1b,
                            const float* __restrict__ fc2w, const float* __restrict__ fc2b,
                            float* __restrict__ out) {
  const int b = blockIdx.x * blockDim.x + threadIdx.x;
  if (b >= B_DIM) return;
  const float* h = hlast + b * H_DIM;
  float z[FC_DIM];
#pragma unroll
  for (int j = 0; j < FC_DIM; ++j) {
    float a = fc1b[j];
    const float* w = fc1w + j * H_DIM;
    for (int k = 0; k < H_DIM; ++k) a = fmaf(w[k], h[k], a);
    z[j] = ftanh(a);
  }
  float l0 = fc2b[0], l1 = fc2b[1];
#pragma unroll
  for (int j = 0; j < FC_DIM; ++j) {
    l0 = fmaf(fc2w[j], z[j], l0);
    l1 = fmaf(fc2w[FC_DIM + j], z[j], l1);
  }
  const float m = fmaxf(l0, l1);
  const float lse = m + logf(__expf(l0 - m) + __expf(l1 - m));
  out[b * 2 + 0] = l0 - lse;
  out[b * 2 + 1] = l1 - lse;
}

// ---------------- host driver ----------------
extern "C" void kernel_launch(void* const* d_in, const int* in_sizes, int n_in,
                              void* d_out, int out_size, void* d_ws, size_t ws_size,
                              hipStream_t stream) {
  const float* x = (const float*)d_in[0];
  const float* wi[4]  = {(const float*)d_in[1], (const float*)d_in[2], (const float*)d_in[3], (const float*)d_in[4]};
  const float* wi2[4] = {(const float*)d_in[5], (const float*)d_in[6], (const float*)d_in[7], (const float*)d_in[8]};
  const float* wh[4]  = {(const float*)d_in[9], (const float*)d_in[10], (const float*)d_in[11], (const float*)d_in[12]};
  const float* bg = (const float*)d_in[13];
  const float* bi = (const float*)d_in[14];
  const float* bf = (const float*)d_in[15];
  const float* bo = (const float*)d_in[16];
  const float* fc1w = (const float*)d_in[17];
  const float* fc1b = (const float*)d_in[18];
  const float* fc2w = (const float*)d_in[19];
  const float* fc2b = (const float*)d_in[20];

  char* ws = (char*)d_ws;
  size_t off = 0;
  auto alloc = [&](size_t bytes) -> void* {
    void* p = ws + off;
    off = (off + bytes + 255) & ~(size_t)255;
    return p;
  };

  const size_t N_X   = (size_t)T_DIM * B_DIM * I_DIM;          // 2M
  const size_t N_WI  = (size_t)B_DIM * H_DIM * I_DIM;          // 4M per gate
  const size_t N_WH  = (size_t)B_DIM * H_DIM * H_DIM;          // 8M per gate

  unsigned short* x_bf   = (unsigned short*)alloc(N_X * 2);
  unsigned short* wih_bf = (unsigned short*)alloc(4 * N_WI * 2);
  unsigned short* wi2_bf = (unsigned short*)alloc(4 * N_WH * 2);
  unsigned short* wh_bf  = (unsigned short*)alloc(4 * N_WH * 2);
  float*          Pbuf   = (float*)alloc(4 * (size_t)B_DIM * T_DIM * H_DIM * 4);
  unsigned short* hs_bf  = (unsigned short*)alloc((size_t)B_DIM * T_DIM * H_DIM * 2);
  float*          hlast  = (float*)alloc((size_t)B_DIM * H_DIM * 4);

  // --- converts ---
  cvt_bf16_kernel<<<1024, 256, 0, stream>>>(x, x_bf, (int)N_X);
  for (int g = 0; g < 4; ++g)
    cvt_bf16_kernel<<<2048, 256, 0, stream>>>(wi[g],  wih_bf + g * N_WI, (int)N_WI);
  for (int g = 0; g < 4; ++g)
    cvt_bf16_kernel<<<4096, 256, 0, stream>>>(wi2[g], wi2_bf + g * N_WH, (int)N_WH);
  for (int g = 0; g < 4; ++g)
    cvt_bf16_kernel<<<4096, 256, 0, stream>>>(wh[g],  wh_bf + g * N_WH, (int)N_WH);

  // --- layer 1: input projections (WMMA bf16, K=I=128), then scan ---
  proj_wmma_kernel<<<2048, 256, 0, stream>>>(
      x_bf, (long long)I_DIM, (long long)B_DIM * I_DIM, wih_bf, I_DIM, Pbuf);
  lstm_scan_kernel<<<B_DIM, 256, 0, stream>>>(Pbuf, wh_bf, bg, bi, bf, bo, hs_bf, hlast);

  // --- layer 2: input projections on hs1 (WMMA bf16, K=H=256), then scan ---
  proj_wmma_kernel<<<2048, 256, 0, stream>>>(
      hs_bf, (long long)T_DIM * H_DIM, (long long)H_DIM, wi2_bf, H_DIM, Pbuf);
  lstm_scan_kernel<<<B_DIM, 256, 0, stream>>>(Pbuf, wh_bf, bg, bi, bf, bo, hs_bf, hlast);

  // --- head ---
  head_kernel<<<1, 128, 0, stream>>>(hlast, fc1w, fc1b, fc2w, fc2b, (float*)d_out);
}
